// GraphConvLayer_69114613730766
// MI455X (gfx1250) — matrix operand order repaired and checked by reading, not compiled, not run on hardware
//
#include <hip/hip_runtime.h>
#include <hip/hip_bf16.h>
#include <math.h>

#define D 128

typedef __attribute__((ext_vector_type(16))) __bf16 v16bf;
typedef __attribute__((ext_vector_type(8)))  __bf16 v8bf;
typedef __attribute__((ext_vector_type(8)))  float  v8f;

__device__ __forceinline__ v16bf cat8(v8bf lo, v8bf hi) {
    return __builtin_shufflevector(lo, hi, 0,1,2,3,4,5,6,7,8,9,10,11,12,13,14,15);
}

// ---------------------------------------------------------------------------
// Zero a float buffer (float4 granularity).
// ---------------------------------------------------------------------------
__global__ void zero4_kernel(float4* __restrict__ p, long long n4) {
    long long i = (long long)blockIdx.x * blockDim.x + threadIdx.x;
    if (i < n4) p[i] = make_float4(0.f, 0.f, 0.f, 0.f);
}

// ---------------------------------------------------------------------------
// Build attention-scaled, transposed bf16 weights WcT[n][k] (128 x 512) and
// fused bias bc[128].  K-blocks: k/128 = t in {0,1,2} -> W_edge[t], 3 -> W_self.
// ---------------------------------------------------------------------------
__global__ void prep_kernel(const float* __restrict__ W_edge,
                            const float* __restrict__ W_self,
                            const float* __restrict__ b_edge,
                            const float* __restrict__ b_self,
                            const float* __restrict__ attn,
                            __bf16* __restrict__ WcT,
                            float* __restrict__ bc) {
    float l0 = attn[0], l1 = attn[1], l2 = attn[2], l3 = attn[3];
    float mx = fmaxf(fmaxf(l0, l1), fmaxf(l2, l3));
    float e0 = __expf(l0 - mx), e1 = __expf(l1 - mx);
    float e2 = __expf(l2 - mx), e3 = __expf(l3 - mx);
    float inv = 1.0f / (e0 + e1 + e2 + e3);
    float a[4] = {e0 * inv, e1 * inv, e2 * inv, e3 * inv};

    int idx = blockIdx.x * blockDim.x + threadIdx.x;   // 0 .. 128*512-1
    if (idx < D * 512) {
        int n  = idx >> 9;
        int k  = idx & 511;
        int t  = k >> 7;
        int kk = k & 127;
        float w = (t < 3) ? W_edge[((size_t)t * D + kk) * D + n]
                          : W_self[(size_t)kk * D + n];
        WcT[(size_t)n * 512 + k] = (__bf16)(w * a[t]);
    }
    if (idx < D) {
        bc[idx] = a[0] * b_edge[idx] + a[1] * b_edge[D + idx] +
                  a[2] * b_edge[2 * D + idx] + a[3] * b_self[idx];
    }
}

// ---------------------------------------------------------------------------
// Scatter (fused): one pass over all edges, accumulator agg3 is (3, N, D).
// ---------------------------------------------------------------------------
__global__ __launch_bounds__(256)
void scatter_all_kernel(const float* __restrict__ x,
                        const long long* __restrict__ edge_index,
                        const int* __restrict__ etype,
                        const float* __restrict__ ew,
                        float* __restrict__ agg3,
                        int nNodes, int nEdges) {
    int e = blockIdx.x * 2 + (threadIdx.x >> 7);
    int j = threadIdx.x & 127;
    if (e >= nEdges) return;
    int t = etype[e];                          // uniform across the wave
    long long s = edge_index[e];
    long long d = edge_index[nEdges + e];
    float w = ew[e];
    atomicAdd(&agg3[((size_t)t * nNodes + (size_t)d) * D + j],
              w * x[(size_t)s * D + j]);
}

// ---------------------------------------------------------------------------
// Scatter (fallback): edges of a single type into a (N, D) accumulator.
// ---------------------------------------------------------------------------
__global__ __launch_bounds__(256)
void scatter_kernel(const float* __restrict__ x,
                    const long long* __restrict__ edge_index,
                    const int* __restrict__ etype,
                    const float* __restrict__ ew,
                    float* __restrict__ agg,
                    int t, int nEdges) {
    int e = blockIdx.x * 2 + (threadIdx.x >> 7);
    int j = threadIdx.x & 127;
    if (e >= nEdges) return;
    if (etype[e] != t) return;
    long long s = edge_index[e];
    long long d = edge_index[nEdges + e];
    float w = ew[e];
    atomicAdd(&agg[(size_t)d * D + j], w * x[(size_t)s * D + j]);
}

// ---------------------------------------------------------------------------
// One K-block (128 wide, 4 steps of 32) of the WMMA GEMM.
// Tile: 128x128 C per block; 8 waves (4M x 2N); acc = 2x4 16x16 f32 frags.
// ---------------------------------------------------------------------------
__device__ __forceinline__
void gemm_kblock(const float* __restrict__ A,        // (nRows, 128) f32
                 const __bf16* __restrict__ WcT,     // (128, 512) bf16
                 int kBlock, int mBase, int nRows,
                 int tid, int lane, int wm, int wn,
                 __bf16 (*As)[40], __bf16 (*Bs)[40],
                 v8f acc[2][4]) {
    const int r  = tid >> 1;                  // tile row / tile n : 0..127
    const int ks = (tid & 1) * 16;            // k half: 0 or 16

    for (int k0 = 0; k0 < 128; k0 += 32) {
        // ---- cooperative A tile load: f32 -> bf16 --------------------------
        {
            int m = mBase + r;
            float vals[16];
            if (m < nRows) {
                const float4* src = (const float4*)(A + (size_t)m * D + k0 + ks);
#pragma unroll
                for (int i = 0; i < 4; ++i) {
                    float4 v = src[i];
                    vals[4 * i + 0] = v.x; vals[4 * i + 1] = v.y;
                    vals[4 * i + 2] = v.z; vals[4 * i + 3] = v.w;
                }
            } else {
#pragma unroll
                for (int i = 0; i < 16; ++i) vals[i] = 0.0f;
            }
#pragma unroll
            for (int i = 0; i < 16; ++i) As[r][ks + i] = (__bf16)vals[i];
        }
        // ---- cooperative B tile load: bf16 copy ----------------------------
        {
            const v8bf* src =
                (const v8bf*)(WcT + (size_t)r * 512 + kBlock * D + k0 + ks);
            *(v8bf*)&Bs[r][ks]     = src[0];
            *(v8bf*)&Bs[r][ks + 8] = src[1];
        }
        __syncthreads();

        // ---- fragment loads per ISA 7.12.2 bf16 layouts --------------------
        const int b8 = (lane >= 16) ? 8  : 0;   // A: K pair offset
        const int bk = (lane >= 16) ? 8  : 0;   // B: K pair offset
        v16bf afrag[2], bfrag[4];
#pragma unroll
        for (int f = 0; f < 2; ++f) {
            int m = wm * 32 + f * 16 + (lane & 15);
            afrag[f] = cat8(*(const v8bf*)&As[m][b8],
                            *(const v8bf*)&As[m][16 + b8]);
        }
#pragma unroll
        for (int g = 0; g < 4; ++g) {
            int n = wn * 64 + g * 16 + (lane & 15);
            bfrag[g] = cat8(*(const v8bf*)&Bs[n][bk],
                            *(const v8bf*)&Bs[n][16 + bk]);
        }
#pragma unroll
        for (int f = 0; f < 2; ++f)
#pragma unroll
            for (int g = 0; g < 4; ++g)
                acc[f][g] = __builtin_amdgcn_wmma_f32_16x16x32_bf16(
                    false, afrag[f], false, bfrag[g],
                    (short)0, acc[f][g], false, false);
        __syncthreads();
    }
}

// mode 0: store  1: accumulate  2: accum + bias + gelu  3: store + bias + gelu
__device__ __forceinline__
void gemm_epilogue(v8f acc[2][4], float* __restrict__ C,
                   const float* __restrict__ bc,
                   int nRows, int mBase, int lane, int wm, int wn, int mode) {
    const int mrow = (lane >= 16) ? 8 : 0;
    const int ncol = lane & 15;
#pragma unroll
    for (int f = 0; f < 2; ++f) {
#pragma unroll
        for (int g = 0; g < 4; ++g) {
            int col = wn * 64 + g * 16 + ncol;
#pragma unroll
            for (int v = 0; v < 8; ++v) {
                int row = mBase + wm * 32 + f * 16 + mrow + v;
                if (row < nRows) {
                    size_t o = (size_t)row * D + col;
                    float val = acc[f][g][v];
                    if (mode == 0) {
                        C[o] = val;
                    } else if (mode == 1) {
                        C[o] += val;
                    } else {
                        float s = val + bc[col] + (mode == 2 ? C[o] : 0.0f);
                        C[o] = 0.5f * s *
                               (1.0f + erff(s * 0.70710678118654752f));
                    }
                }
            }
        }
    }
}

// ---------------------------------------------------------------------------
// Fused GEMM: K = 512 over [agg3 type 0..2 | x], single C write w/ bias+GELU.
// ---------------------------------------------------------------------------
__global__ __launch_bounds__(256)
void gemm_fused_kernel(const float* __restrict__ agg3,
                       const float* __restrict__ x,
                       const __bf16* __restrict__ WcT,
                       const float* __restrict__ bc,
                       float* __restrict__ C, int nRows) {
    __shared__ __align__(16) __bf16 As[128][40];
    __shared__ __align__(16) __bf16 Bs[128][40];

    const int tid  = threadIdx.x;
    const int lane = tid & 31;
    const int wave = tid >> 5;
    const int wm   = wave >> 1;
    const int wn   = wave & 1;
    const int mBase = blockIdx.x * 128;

    v8f acc[2][4];
#pragma unroll
    for (int f = 0; f < 2; ++f)
#pragma unroll
        for (int g = 0; g < 4; ++g)
#pragma unroll
            for (int i = 0; i < 8; ++i) acc[f][g][i] = 0.0f;

    const size_t typeStride = (size_t)nRows * D;
    for (int kb = 0; kb < 4; ++kb) {
        const float* A = (kb < 3) ? (agg3 + (size_t)kb * typeStride) : x;
        gemm_kblock(A, WcT, kb, mBase, nRows, tid, lane, wm, wn, As, Bs, acc);
    }
    gemm_epilogue(acc, C, bc, nRows, mBase, lane, wm, wn, /*mode=*/3);
}

// ---------------------------------------------------------------------------
// Per-type GEMM (fallback path, small workspace).
// ---------------------------------------------------------------------------
__global__ __launch_bounds__(256)
void gemm_wmma_kernel(const float* __restrict__ A,
                      const __bf16* __restrict__ WcT,
                      const float* __restrict__ bc,
                      float* __restrict__ C,
                      int nRows, int kBlock, int mode) {
    __shared__ __align__(16) __bf16 As[128][40];
    __shared__ __align__(16) __bf16 Bs[128][40];

    const int tid  = threadIdx.x;
    const int lane = tid & 31;
    const int wave = tid >> 5;
    const int wm   = wave >> 1;
    const int wn   = wave & 1;
    const int mBase = blockIdx.x * 128;

    v8f acc[2][4];
#pragma unroll
    for (int f = 0; f < 2; ++f)
#pragma unroll
        for (int g = 0; g < 4; ++g)
#pragma unroll
            for (int i = 0; i < 8; ++i) acc[f][g][i] = 0.0f;

    gemm_kblock(A, WcT, kBlock, mBase, nRows, tid, lane, wm, wn, As, Bs, acc);
    gemm_epilogue(acc, C, bc, nRows, mBase, lane, wm, wn, mode);
}

// ---------------------------------------------------------------------------
extern "C" void kernel_launch(void* const* d_in, const int* in_sizes, int n_in,
                              void* d_out, int out_size, void* d_ws, size_t ws_size,
                              hipStream_t stream) {
    const float*     x      = (const float*)d_in[0];
    const long long* ei     = (const long long*)d_in[1];
    const int*       etype  = (const int*)d_in[2];
    const float*     ew     = (const float*)d_in[3];
    const float*     W_edge = (const float*)d_in[4];
    const float*     b_edge = (const float*)d_in[5];
    const float*     W_self = (const float*)d_in[6];
    const float*     b_self = (const float*)d_in[7];
    const float*     attn   = (const float*)d_in[8];
    float*           out    = (float*)d_out;

    const int nNodes = in_sizes[0] / D;     // 100000
    const int nEdges = in_sizes[2];         // 600000
    const int mTiles = (nNodes + 127) / 128;

    char* ws = (char*)d_ws;
    const size_t agg1Bytes = (((size_t)nNodes * D * sizeof(float)) + 255) & ~(size_t)255;
    const size_t agg3Bytes = ((3ull * nNodes * D * sizeof(float)) + 255) & ~(size_t)255;
    const size_t wctBytes  = (((size_t)D * 512 * sizeof(__bf16)) + 255) & ~(size_t)255;
    const size_t bcBytes   = 512;

    const bool big = ws_size >= agg3Bytes + wctBytes + bcBytes;

    if (big) {
        // ---- fused path: one scatter pass, one K=512 GEMM, single C write --
        float*  agg3 = (float*)ws;
        __bf16* WcT  = (__bf16*)(ws + agg3Bytes);
        float*  bc   = (float*)(ws + agg3Bytes + wctBytes);

        prep_kernel<<<(D * 512 + 255) / 256, 256, 0, stream>>>(
            W_edge, W_self, b_edge, b_self, attn, WcT, bc);

        long long n4 = 3ll * nNodes * D / 4;
        zero4_kernel<<<(unsigned)((n4 + 255) / 256), 256, 0, stream>>>(
            (float4*)agg3, n4);
        scatter_all_kernel<<<(nEdges + 1) / 2, 256, 0, stream>>>(
            x, ei, etype, ew, agg3, nNodes, nEdges);
        gemm_fused_kernel<<<mTiles, 256, 0, stream>>>(
            agg3, x, WcT, bc, out, nNodes);
    } else {
        // ---- fallback: per-type passes with 51MB accumulator ---------------
        float*  agg = (float*)ws;
        __bf16* WcT = (__bf16*)(ws + agg1Bytes);
        float*  bc  = (float*)(ws + agg1Bytes + wctBytes);

        prep_kernel<<<(D * 512 + 255) / 256, 256, 0, stream>>>(
            W_edge, W_self, b_edge, b_self, attn, WcT, bc);

        long long n4 = (long long)nNodes * D / 4;
        for (int t = 0; t < 3; ++t) {
            zero4_kernel<<<(unsigned)((n4 + 255) / 256), 256, 0, stream>>>(
                (float4*)agg, n4);
            scatter_kernel<<<(nEdges + 1) / 2, 256, 0, stream>>>(
                x, ei, etype, ew, agg, t, nEdges);
            gemm_wmma_kernel<<<mTiles, 256, 0, stream>>>(
                agg, WcT, bc, out, nNodes, t, (t == 0) ? 0 : 1);
        }
        gemm_wmma_kernel<<<mTiles, 256, 0, stream>>>(
            x, WcT, bc, out, nNodes, 3, 2);
    }
}
